// FermiNet_46858093199812
// MI455X (gfx1250) — matrix operand order repaired and checked by reading, not compiled, not run on hardware
//
#include <hip/hip_runtime.h>
#include <hip/hip_bf16.h>

// ---------------------------------------------------------------------------
// FermiNet forward: B=2048, NE=32, NO=32, H0=H1=256, H2=64
// Dominant cost: einsum bnh,ohk->bnok == GEMM (65536 x 2048 x 256) ~ 68.7 GFLOP
// f16 WMMA (v_wmma_f32_16x16x32_f16) for both 256-K GEMMs, software-pipelined
// with sched_barrier-enforced double buffering so next-step loads are issued
// BEFORE the current WMMA burst (loads in flight across every WMMA).
// Fused tanh/Wo2 epilogue keeps the 256MB `h` tensor out of memory entirely.
// ---------------------------------------------------------------------------

typedef __attribute__((ext_vector_type(16))) _Float16 v16h;
typedef __attribute__((ext_vector_type(8)))  _Float16 v8h;
typedef __attribute__((ext_vector_type(8)))  float    v8f;

#define H0     256
#define H1     256
#define H2     64
#define NO     32

__device__ __forceinline__ float fast_tanh(float x) {
    float cx = fminf(fmaxf(x, -15.f), 15.f);
    float e  = __expf(2.f * cx);
    return (e - 1.f) / (e + 1.f);
}

// A-fragment: 16x32 (MxK) f16 from row-major source (rowp -> this lane's row).
// lane 0-15 : K = k0+[0..7], k0+[16..23]; lane 16-31: K = k0+[8..15], k0+[24..31]
__device__ __forceinline__ v16h load_a_frag(const _Float16* rowp, int k0, int lane) {
    union { v16h v; v8h h[2]; } u;
    const _Float16* p = rowp + k0 + ((lane >> 4) << 3);
    u.h[0] = *(const v8h*)(p);
    u.h[1] = *(const v8h*)(p + 16);
    return u.v;
}

// B-fragment: 32x16 (KxN) f16 from N-major (transposed) source (colp -> col n).
// lane 0-15: K=k0+[0..7], k0+[16..23]; lane 16-31: K=k0+[8..15], k0+[24..31].
__device__ __forceinline__ v16h load_b_frag(const _Float16* colp, int k0, int lane) {
    union { v16h v; v8h h[2]; } u;
    const _Float16* p = colp + k0 + ((lane >> 4) << 3);
    u.h[0] = *(const v8h*)(p);
    u.h[1] = *(const v8h*)(p + 16);
    return u.v;
}

__device__ __forceinline__ v8f wmma_f16(v16h a, v16h b, v8f c) {
    return __builtin_amdgcn_wmma_f32_16x16x32_f16(false, a, false, b, (short)0, c, false, false);
}

// --------------------------- weight prep ----------------------------------
__global__ void __launch_bounds__(256) k_tr_we2(const float* __restrict__ We2,
                                                _Float16* __restrict__ we2t) {
    int idx = blockIdx.x * 256 + threadIdx.x;          // 65536
    int n = idx >> 8, k = idx & 255;
    we2t[idx] = (_Float16)We2[k * 256 + n];
}

__global__ void __launch_bounds__(256) k_tr_wo1(const float* __restrict__ Wo1,
                                                _Float16* __restrict__ wo1t) {
    int idx = blockIdx.x * 256 + threadIdx.x;          // 524288
    int o = idx >> 14;
    int rem = idx & 16383;
    int n = rem >> 8, k = rem & 255;
    wo1t[idx] = (_Float16)Wo1[o * 16384 + k * 64 + n];
}

// --------------------------- layer 1 (K=3) --------------------------------
__global__ void __launch_bounds__(256) k_embed1(const float* __restrict__ x,
                                                const float* __restrict__ We1,
                                                const float* __restrict__ be1,
                                                _Float16* __restrict__ g) {
    int idx = blockIdx.x * 256 + threadIdx.x;          // ROWS*256
    int r = idx >> 8, c = idx & 255;
    const float* xr = x + (size_t)r * 3;
    float v = fmaf(xr[0], We1[c],
              fmaf(xr[1], We1[256 + c],
              fmaf(xr[2], We1[512 + c], be1[c])));
    g[idx] = (_Float16)fast_tanh(v);
}

// ------------------- layer 2 GEMM: f = tanh(g @ We2 + be2) -----------------
// block = 256 threads = 8 waves; wave (mt,nt) computes a 16x64 tile.
// sched_barrier-pinned pipeline: next-step loads issue before current WMMAs.
__global__ void __launch_bounds__(256) k_gemm_f(const _Float16* __restrict__ g,
                                                const _Float16* __restrict__ we2t,
                                                const float* __restrict__ be2,
                                                _Float16* __restrict__ f) {
    int lane = threadIdx.x & 31;
    int wave = threadIdx.x >> 5;
    int mt = wave >> 2, nt = wave & 3;
    int row0 = blockIdx.x * 32 + mt * 16;
    int col0 = nt * 64;
    const _Float16* arow = g + (size_t)(row0 + (lane & 15)) * H0;
    const _Float16* bcol = we2t + (size_t)(col0 + (lane & 15)) * H0;

    v8f acc[4] = {};
    v16h a = load_a_frag(arow, 0, lane);
    v16h bf[4], bn[4];
#pragma unroll
    for (int s = 0; s < 4; ++s) bf[s] = load_b_frag(bcol + s * 16 * H0, 0, lane);

#pragma unroll
    for (int k0 = 0; k0 < H0; k0 += 32) {
        v16h an = a;
        if (k0 + 32 < H0) {                      // compile-time folded per-iter
            an = load_a_frag(arow, k0 + 32, lane);
#pragma unroll
            for (int s = 0; s < 4; ++s)
                bn[s] = load_b_frag(bcol + s * 16 * H0, k0 + 32, lane);
        }
        __builtin_amdgcn_sched_barrier(0);       // loads above stay above
#pragma unroll
        for (int s = 0; s < 4; ++s)
            acc[s] = wmma_f16(a, bf[s], acc[s]);
        __builtin_amdgcn_sched_barrier(0);       // wmma burst stays together
        a = an;
#pragma unroll
        for (int s = 0; s < 4; ++s) bf[s] = bn[s];
    }

    int rbase = row0 + ((lane >> 4) << 3);
#pragma unroll
    for (int s = 0; s < 4; ++s) {
        int col = col0 + s * 16 + (lane & 15);
        float bias = be2[col];
#pragma unroll
        for (int r = 0; r < 8; ++r) {
            float v = fast_tanh(acc[s][r] + bias);
            f[(size_t)(rbase + r) * H1 + col] = (_Float16)v;
        }
    }
}

// ------ fused orbital: M[row][o] = sum_k tanh(f@Wo1[o]+bo1[o])_k * Wo2[o][k] + bo2[o]
// block = 256 threads = 8 waves; block handles 16 rows x all 32 orbitals.
// A comes from LDS (staged once), B pipelined from global (L2-hot 1MB).
__global__ void __launch_bounds__(256) k_orbital(const _Float16* __restrict__ f,
                                                 const _Float16* __restrict__ wo1t,
                                                 const float* __restrict__ bo1,
                                                 const float* __restrict__ wo2,
                                                 const float* __restrict__ bo2,
                                                 float* __restrict__ Mout) {
    __shared__ __align__(16) _Float16 sf[16 * H1];     // 8 KB f-tile
    int rt   = blockIdx.x;                             // 0..4095 (16-row tiles)
    int tid  = threadIdx.x;
    {
        const float4* src = (const float4*)(f + (size_t)rt * 16 * H1);
        float4* dst = (float4*)sf;
        dst[tid]       = src[tid];
        dst[tid + 256] = src[tid + 256];
    }
    __syncthreads();

    int lane = tid & 31, wave = tid >> 5;
    const _Float16* arow = sf + (lane & 15) * H1;
    int row0 = rt * 16 + ((lane >> 4) << 3);

    for (int oi = 0; oi < 4; ++oi) {
        int o = wave * 4 + oi;
        const _Float16* bcol = wo1t + (size_t)o * H2 * H1 + (size_t)(lane & 15) * H1;

        v8f acc[4] = {};
        v16h bf[4], bn[4];
#pragma unroll
        for (int s = 0; s < 4; ++s) bf[s] = load_b_frag(bcol + s * 16 * H1, 0, lane);

#pragma unroll
        for (int k0 = 0; k0 < H1; k0 += 32) {
            if (k0 + 32 < H1) {
#pragma unroll
                for (int s = 0; s < 4; ++s)
                    bn[s] = load_b_frag(bcol + s * 16 * H1, k0 + 32, lane);
            }
            v16h a = load_a_frag(arow, k0, lane);      // LDS: low latency
            __builtin_amdgcn_sched_barrier(0);         // keep prefetch above
#pragma unroll
            for (int s = 0; s < 4; ++s)
                acc[s] = wmma_f16(a, bf[s], acc[s]);
            __builtin_amdgcn_sched_barrier(0);
#pragma unroll
            for (int s = 0; s < 4; ++s) bf[s] = bn[s];
        }

        // epilogue: tanh + dot with Wo2 over n (64 = 4 subtiles x 16 lanes)
        float s_r[8];
#pragma unroll
        for (int r = 0; r < 8; ++r) s_r[r] = 0.f;
#pragma unroll
        for (int s = 0; s < 4; ++s) {
            int n = s * 16 + (lane & 15);
            float b1 = bo1[o * H2 + n];
            float w2 = wo2[o * H2 + n];
#pragma unroll
            for (int r = 0; r < 8; ++r)
                s_r[r] += fast_tanh(acc[s][r] + b1) * w2;
        }
        float b2 = bo2[o];
#pragma unroll
        for (int r = 0; r < 8; ++r) {
            float v = s_r[r];
            v += __shfl_xor(v, 1);
            v += __shfl_xor(v, 2);
            v += __shfl_xor(v, 4);
            v += __shfl_xor(v, 8);     // reduce across the 16-lane N group
            if ((lane & 15) == 0)
                Mout[(size_t)(row0 + r) * NO + o] = v + b2;
        }
    }
}

// --------------------- slogdet of 2048 32x32 matrices ----------------------
// one wave32 per matrix; LU with partial pivoting in LDS; sum log|U_kk|.
__global__ void __launch_bounds__(32) k_slogdet(const float* __restrict__ Mb,
                                                float* __restrict__ out) {
    __shared__ float A[32][33];
    int b = blockIdx.x;
    int lane = threadIdx.x;
    const float* src = Mb + (size_t)b * 1024;
#pragma unroll
    for (int j = 0; j < 32; ++j) A[lane][j] = src[lane * 32 + j];
    __syncthreads();

    float logdet = 0.f;
    for (int k = 0; k < 32; ++k) {
        // argmax |A[i][k]| over i >= k (deterministic tie-break: lowest index)
        float v = (lane >= k) ? fabsf(A[lane][k]) : -1.f;
        int idx = lane;
        for (int m = 16; m > 0; m >>= 1) {
            float ov = __shfl_xor(v, m);
            int   oi = __shfl_xor(idx, m);
            if (ov > v || (ov == v && oi < idx)) { v = ov; idx = oi; }
        }
        int p = idx;
        if (p != k) {                      // swap rows k,p; lane = column
            float t = A[k][lane];
            A[k][lane] = A[p][lane];
            A[p][lane] = t;
        }
        __syncthreads();
        float piv = A[k][k];
        logdet += __logf(fabsf(piv));
        float akj = A[k][lane];            // lane = column j
        for (int i = k + 1; i < 32; ++i) {
            float m_ = A[i][k] / piv;      // broadcast read precedes column write
            A[i][lane] -= m_ * akj;
        }
        __syncthreads();
    }
    if (lane == 0) out[b] = logdet;
}

// ---------------------------------------------------------------------------
extern "C" void kernel_launch(void* const* d_in, const int* in_sizes, int n_in,
                              void* d_out, int out_size, void* d_ws, size_t ws_size,
                              hipStream_t stream) {
    const float* x   = (const float*)d_in[0];
    const float* We1 = (const float*)d_in[1];
    const float* be1 = (const float*)d_in[2];
    const float* We2 = (const float*)d_in[3];
    const float* be2 = (const float*)d_in[4];
    const float* Wo1 = (const float*)d_in[5];
    const float* bo1 = (const float*)d_in[6];
    const float* Wo2 = (const float*)d_in[7];
    const float* bo2 = (const float*)d_in[8];
    float* out = (float*)d_out;

    char* ws = (char*)d_ws;
    _Float16* we2t = (_Float16*)(ws);                      // 256*256*2      = 128 KB
    _Float16* wo1t = (_Float16*)(ws + 131072);             // 32*64*256*2    = 1 MB
    _Float16* g    = (_Float16*)(ws + 1179648);            // 65536*256*2    = 32 MB
    _Float16* fbuf = (_Float16*)(ws + 34734080);           // 65536*256*2    = 32 MB
    float*    Mbuf = (float*)  (ws + 68288512);            // 2048*32*32*4   = 8 MB

    k_tr_we2 <<<256,   256, 0, stream>>>(We2, we2t);
    k_tr_wo1 <<<2048,  256, 0, stream>>>(Wo1, wo1t);
    k_embed1 <<<65536, 256, 0, stream>>>(x, We1, be1, g);
    k_gemm_f <<<2048,  256, 0, stream>>>(g, we2t, be2, fbuf);
    k_orbital<<<4096,  256, 0, stream>>>(fbuf, wo1t, bo1, Wo2, bo2, Mbuf);
    k_slogdet<<<2048,  32,  0, stream>>>(Mbuf, out);
}
